// DAPMSALayer_36713380446453
// MI455X (gfx1250) — matrix operand order repaired and checked by reading, not compiled, not run on hardware
//
#include <hip/hip_runtime.h>

#define BN 384
#define SS 256
#define MS 64
#define ZD 128
#define HH 8
#define CH 32
#define HC 256
#define COPM 32

typedef __attribute__((ext_vector_type(16))) __bf16 v16bf;
typedef __attribute__((ext_vector_type(8)))  __bf16 v8bf;
typedef __attribute__((ext_vector_type(8)))  float  v8f;
typedef __attribute__((ext_vector_type(8)))  unsigned short us8;

__device__ __forceinline__ unsigned short f2bfu(float f) {
  unsigned u = __builtin_bit_cast(unsigned, f);
  unsigned r = (u + 0x7FFFu + ((u >> 16) & 1u)) >> 16;
  return (unsigned short)r;
}
__device__ __forceinline__ float bf2f(unsigned short s) {
  unsigned u = ((unsigned)s) << 16;
  return __builtin_bit_cast(float, u);
}
__device__ __forceinline__ float fast_sigmoid(float x) {
  return __builtin_amdgcn_rcpf(1.0f + __expf(-x));   // v_exp + v_rcp, no IEEE divide
}
__device__ __forceinline__ v8bf ld8(const unsigned short* p) {
  us8 t = *(const us8*)p;
  return __builtin_bit_cast(v8bf, t);
}
// A fragment (16xK tile, 16-bit): lane<16 holds K={kb..kb+7, kb+16..kb+23}, lane>=16 shifted by 8
__device__ __forceinline__ v16bf load_fragA(const unsigned short* base, int rowStride,
                                            int m16, int kb, int lane) {
  int m  = m16 + (lane & 15);
  int k0 = kb + ((lane >> 4) << 3);
  const unsigned short* p = base + (long)m * rowStride + k0;
  v8bf lo = ld8(p), hi = ld8(p + 16);
  return __builtin_shufflevector(lo, hi, 0,1,2,3,4,5,6,7,8,9,10,11,12,13,14,15);
}
// B fragment (Kx16 tile): lane holds column n, K={kb..kb+7, kb+16..kb+23}, lane>=16 shifted by 8
__device__ __forceinline__ v16bf load_fragB(const unsigned short* base, int rowStride,
                                            int n16, int kb, int lane) {
  int n  = n16 + (lane & 15);
  int k0 = kb + ((lane >> 4) << 3);
  const unsigned short* p = base + (long)n * rowStride + k0;
  v8bf lo = ld8(p), hi = ld8(p + 16);
  return __builtin_shufflevector(lo, hi, 0,1,2,3,4,5,6,7,8,9,10,11,12,13,14,15);
}
__device__ __forceinline__ v8f wmma_bf(v16bf a, v16bf b, v8f c) {
  return __builtin_amdgcn_wmma_f32_16x16x32_bf16(false, a, false, b, (short)0, c, false, false);
}

// ---------------- weight convert ----------------
__global__ void k_cvt(const float* __restrict__ src, unsigned short* __restrict__ dst, int n) {
  int i = blockIdx.x * blockDim.x + threadIdx.x;
  if (i < n) dst[i] = f2bfu(src[i]);
}

// ---------------- pair bias: ln(z) @ proj_z_w.T + mask ----------------
__global__ void k_bias(const float* __restrict__ z, const float* __restrict__ tmask,
                       const float* __restrict__ nw, const float* __restrict__ nb,
                       const float* __restrict__ pw, float* __restrict__ wbias) {
  int ij = blockIdx.x;           // i*BN + j
  int i = ij / BN, j = ij % BN;
  int t = threadIdx.x;           // 128 threads == ZD
  __shared__ float red[128];
  float x = z[(long)ij * ZD + t];
  red[t] = x; __syncthreads();
  for (int s = 64; s > 0; s >>= 1) { if (t < s) red[t] += red[t + s]; __syncthreads(); }
  float mu = red[0] * (1.0f / ZD); __syncthreads();
  float d = x - mu;
  red[t] = d * d; __syncthreads();
  for (int s = 64; s > 0; s >>= 1) { if (t < s) red[t] += red[t + s]; __syncthreads(); }
  float var = red[0] * (1.0f / ZD); __syncthreads();
  float xn = d * rsqrtf(var + 1e-5f) * nw[t] + nb[t];
  float mterm = (1.0f - tmask[ij]) * (-1e9f);
  for (int h = 0; h < HH; h++) {
    red[t] = xn * pw[h * ZD + t]; __syncthreads();
    for (int s = 64; s > 0; s >>= 1) { if (t < s) red[t] += red[t + s]; __syncthreads(); }
    if (t == 0) wbias[((long)h * BN + i) * BN + j] = red[0] + mterm;
    __syncthreads();
  }
}

// ---------------- softmax over j ----------------
__global__ void k_softmax(float* __restrict__ wbias, unsigned short* __restrict__ wbf) {
  long row = blockIdx.x;         // H*BN rows
  float* p = wbias + row * BN;
  int t = threadIdx.x;           // 128
  __shared__ float red[128];
  float mx = -1e30f;
  for (int j = t; j < BN; j += 128) mx = fmaxf(mx, p[j]);
  red[t] = mx; __syncthreads();
  for (int s = 64; s > 0; s >>= 1) { if (t < s) red[t] = fmaxf(red[t], red[t + s]); __syncthreads(); }
  mx = red[0]; __syncthreads();
  float sum = 0.f;
  for (int j = t; j < BN; j += 128) { float e = __expf(p[j] - mx); p[j] = e; sum += e; }
  red[t] = sum; __syncthreads();
  for (int s = 64; s > 0; s >>= 1) { if (t < s) red[t] += red[t + s]; __syncthreads(); }
  float inv = __builtin_amdgcn_rcpf(red[0]);
  for (int j = t; j < BN; j += 128) { float v = p[j] * inv; p[j] = v; wbf[row * BN + j] = f2bfu(v); }
}

// ---------------- ln(m) -> v (bf16 [h][s*32+d][j]) and g (bf16 [s*BN+n][256]) ----------------
// De-diverged: each wave does 4 v-tiles then 4 g-tiles, fully uniform.
__global__ void k2_mvg(const float* __restrict__ m, const float* __restrict__ nw,
                       const float* __restrict__ nb, const unsigned short* __restrict__ wmbf,
                       const unsigned short* __restrict__ wgbf,
                       unsigned short* __restrict__ vbf, unsigned short* __restrict__ gbf) {
  __shared__ unsigned short mnbf[16 * 64];
  int rb = blockIdx.x * 16;      // 16 rows of (s,n), same s (16 | 384)
  int s = rb / BN, n0 = rb % BN;
  int t = threadIdx.x;           // 128 = 4 waves
  if (t < 16) {
    const float* row = m + (long)(rb + t) * MS;
    float mu = 0.f;
    for (int k = 0; k < MS; k++) mu += row[k];
    mu *= (1.0f / MS);
    float var = 0.f;
    for (int k = 0; k < MS; k++) { float d = row[k] - mu; var += d * d; }
    float inv = rsqrtf(var * (1.0f / MS) + 1e-5f);
    for (int k = 0; k < MS; k++) mnbf[t * 64 + k] = f2bfu((row[k] - mu) * inv * nw[k] + nb[k]);
  }
  __syncthreads();
  int wave = t >> 5, lane = t & 31, mh = (lane >> 4) * 8;
  v16bf a0 = load_fragA(mnbf, 64, 0, 0, lane);
  v16bf a1 = load_fragA(mnbf, 64, 0, 32, lane);
#pragma unroll
  for (int q = 0; q < 4; q++) {            // v tiles
    int nt = wave * 4 + q;
    v8f acc = {};
    acc = wmma_bf(a0, load_fragB(wmbf, 64, nt * 16, 0, lane), acc);
    acc = wmma_bf(a1, load_fragB(wmbf, 64, nt * 16, 32, lane), acc);
    int ncol = nt * 16 + (lane & 15);
    int h = ncol >> 5, dd = ncol & 31;
#pragma unroll
    for (int r = 0; r < 8; r++) {
      int n = n0 + mh + r;
      vbf[(((long)h * SS + s) * CH + dd) * BN + n] = f2bfu(acc[r]);
    }
  }
#pragma unroll
  for (int q = 0; q < 4; q++) {            // g tiles
    int nt = wave * 4 + q;
    v8f acc = {};
    acc = wmma_bf(a0, load_fragB(wgbf, 64, nt * 16, 0, lane), acc);
    acc = wmma_bf(a1, load_fragB(wgbf, 64, nt * 16, 32, lane), acc);
    int ncol = nt * 16 + (lane & 15);
#pragma unroll
    for (int r = 0; r < 8; r++) {
      int mrow = mh + r;
      gbf[((long)(rb + mrow)) * HC + ncol] = f2bfu(fast_sigmoid(acc[r]));
    }
  }
}

// ---------------- 8 GEMMs: o = w[h] @ v[h]; gated -> go bf16 [s*BN+i][256] ----------------
// Software-pipelined: next K-step fragments are loaded before current WMMAs issue.
__global__ void k3_pwa(const unsigned short* __restrict__ wbf, const unsigned short* __restrict__ vbf,
                       const unsigned short* __restrict__ gbf, unsigned short* __restrict__ gobf) {
  int wave = threadIdx.x >> 5, lane = threadIdx.x & 31;
  long job = (long)blockIdx.x * 8 + wave;           // 8*24*128 jobs
  int h = (int)(job / (24 * 128));
  int rem = (int)(job % (24 * 128));
  int mtile = rem / 128, ng4 = rem % 128;
  const unsigned short* A  = wbf + (long)h * BN * BN;       // [i][j], K=j contiguous
  const unsigned short* Bm = vbf + (long)h * SS * CH * BN;  // [s*32+d][j]
  int nb0 = ng4 * 4 * 16;
  v8f acc[4] = {};
  v16bf a = load_fragA(A, BN, mtile * 16, 0, lane);
  v16bf b[4];
#pragma unroll
  for (int u = 0; u < 4; u++) b[u] = load_fragB(Bm, BN, nb0 + u * 16, 0, lane);
  for (int kk = 32; kk < BN; kk += 32) {
    v16bf an = load_fragA(A, BN, mtile * 16, kk, lane);
    v16bf bn[4];
#pragma unroll
    for (int u = 0; u < 4; u++) bn[u] = load_fragB(Bm, BN, nb0 + u * 16, kk, lane);
#pragma unroll
    for (int u = 0; u < 4; u++) acc[u] = wmma_bf(a, b[u], acc[u]);
    a = an;
#pragma unroll
    for (int u = 0; u < 4; u++) b[u] = bn[u];
  }
#pragma unroll
  for (int u = 0; u < 4; u++) acc[u] = wmma_bf(a, b[u], acc[u]);

  int mh = (lane >> 4) * 8;
#pragma unroll
  for (int u = 0; u < 4; u++) {
    int ng = nb0 + u * 16 + (lane & 15);
    int s = ng >> 5, d = ng & 31;
#pragma unroll
    for (int r = 0; r < 8; r++) {
      int i = mtile * 16 + mh + r;
      long idx = ((long)(s * BN + i)) * HC + h * CH + d;
      float g = bf2f(gbf[idx]);
      gobf[idx] = f2bfu(acc[u][r] * g);
    }
  }
}

// ---------------- m1 = m + go @ pwa_proj_o_w.T ----------------
__global__ void k4_m1(const float* __restrict__ m, const unsigned short* __restrict__ gobf,
                      const unsigned short* __restrict__ wobf, float* __restrict__ m1) {
  int rb = blockIdx.x * 16;
  int wave = threadIdx.x >> 5, lane = threadIdx.x & 31, mh = (lane >> 4) * 8;
  const unsigned short* A = gobf + (long)rb * HC;
  v8f acc = {};
  v16bf a = load_fragA(A, HC, 0, 0, lane);
  v16bf b = load_fragB(wobf, HC, wave * 16, 0, lane);
  for (int kk = 32; kk < HC; kk += 32) {
    v16bf an = load_fragA(A, HC, 0, kk, lane);
    v16bf bn = load_fragB(wobf, HC, wave * 16, kk, lane);
    acc = wmma_bf(a, b, acc);
    a = an; b = bn;
  }
  acc = wmma_bf(a, b, acc);
  int col = wave * 16 + (lane & 15);
#pragma unroll
  for (int r = 0; r < 8; r++) {
    long row = rb + mh + r;
    m1[row * MS + col] = m[row * MS + col] + acc[r];
  }
}

// ---------------- transition: m2 = m1 + (silu(tn@fc1)* (tn@fc2)) @ fc3.T ----------------
__global__ void k5_trans(const float* __restrict__ m1, const float* __restrict__ nw,
                         const float* __restrict__ nb, const unsigned short* __restrict__ fc1,
                         const unsigned short* __restrict__ fc2, const unsigned short* __restrict__ fc3,
                         float* __restrict__ mout) {
  __shared__ unsigned short tnbf[16 * 64];
  __shared__ unsigned short ubf[16 * 256];
  int rb = blockIdx.x * 16;
  int t = threadIdx.x;           // 128
  if (t < 16) {
    const float* row = m1 + (long)(rb + t) * MS;
    float mu = 0.f;
    for (int k = 0; k < MS; k++) mu += row[k];
    mu *= (1.0f / MS);
    float var = 0.f;
    for (int k = 0; k < MS; k++) { float d = row[k] - mu; var += d * d; }
    float inv = rsqrtf(var * (1.0f / MS) + 1e-5f);
    for (int k = 0; k < MS; k++) tnbf[t * 64 + k] = f2bfu((row[k] - mu) * inv * nw[k] + nb[k]);
  }
  __syncthreads();
  int wave = t >> 5, lane = t & 31, mh = (lane >> 4) * 8;
  v16bf a0 = load_fragA(tnbf, 64, 0, 0, lane);
  v16bf a1 = load_fragA(tnbf, 64, 0, 32, lane);
#pragma unroll
  for (int q = 0; q < 4; q++) {
    int nt = wave * 4 + q;
    v8f x1 = {}, x2 = {};
    x1 = wmma_bf(a0, load_fragB(fc1, 64, nt * 16, 0, lane), x1);
    x1 = wmma_bf(a1, load_fragB(fc1, 64, nt * 16, 32, lane), x1);
    x2 = wmma_bf(a0, load_fragB(fc2, 64, nt * 16, 0, lane), x2);
    x2 = wmma_bf(a1, load_fragB(fc2, 64, nt * 16, 32, lane), x2);
    int col = nt * 16 + (lane & 15);
#pragma unroll
    for (int r = 0; r < 8; r++) {
      float x = x1[r];
      float sil = x * fast_sigmoid(x);
      ubf[(mh + r) * 256 + col] = f2bfu(sil * x2[r]);
    }
  }
  __syncthreads();
  v8f acc = {};
  v16bf a = load_fragA(ubf, 256, 0, 0, lane);
  v16bf b = load_fragB(fc3, HC, wave * 16, 0, lane);
  for (int kk = 32; kk < HC; kk += 32) {
    v16bf an = load_fragA(ubf, 256, 0, kk, lane);
    v16bf bn = load_fragB(fc3, HC, wave * 16, kk, lane);
    acc = wmma_bf(a, b, acc);
    a = an; b = bn;
  }
  acc = wmma_bf(a, b, acc);
  int col = wave * 16 + (lane & 15);
#pragma unroll
  for (int r = 0; r < 8; r++) {
    long row = rb + mh + r;
    mout[row * MS + col] = m1[row * MS + col] + acc[r];
  }
}

// ---------------- OPM a/b projections, stored K(=s)-major: abuf [n][c][s] ----------------
__global__ void k6_ab(const float* __restrict__ m2, const float* __restrict__ msk,
                      const float* __restrict__ nw, const float* __restrict__ nb,
                      const unsigned short* __restrict__ wa, const unsigned short* __restrict__ wb,
                      unsigned short* __restrict__ abuf, unsigned short* __restrict__ bbuf) {
  __shared__ unsigned short onbf[16 * 64];
  int rb = blockIdx.x * 16;
  int s = rb / BN, n0 = rb % BN;
  int t = threadIdx.x;           // 64 = 2 waves
  if (t < 16) {
    const float* row = m2 + (long)(rb + t) * MS;
    float mu = 0.f;
    for (int k = 0; k < MS; k++) mu += row[k];
    mu *= (1.0f / MS);
    float var = 0.f;
    for (int k = 0; k < MS; k++) { float d = row[k] - mu; var += d * d; }
    float inv = rsqrtf(var * (1.0f / MS) + 1e-5f);
    for (int k = 0; k < MS; k++) onbf[t * 64 + k] = f2bfu((row[k] - mu) * inv * nw[k] + nb[k]);
  }
  __syncthreads();
  int wave = t >> 5, lane = t & 31, mh = (lane >> 4) * 8;
  const unsigned short* w = wave ? wb : wa;       // [32][64]
  unsigned short* obuf = wave ? bbuf : abuf;
  v16bf a0 = load_fragA(onbf, 64, 0, 0, lane);
  v16bf a1 = load_fragA(onbf, 64, 0, 32, lane);
#pragma unroll
  for (int nt = 0; nt < 2; nt++) {
    v8f acc = {};
    acc = wmma_bf(a0, load_fragB(w, 64, nt * 16, 0, lane), acc);
    acc = wmma_bf(a1, load_fragB(w, 64, nt * 16, 32, lane), acc);
    int c = nt * 16 + (lane & 15);
#pragma unroll
    for (int r = 0; r < 8; r++) {
      int n = n0 + mh + r;
      float mv = msk[s * BN + n];
      obuf[((long)n * COPM + c) * SS + s] = f2bfu(acc[r] * mv);
    }
  }
}

// ---------------- fused OPM GEMM + 1024->128 projection + residual ----------------
__global__ void k7_opm(const float* __restrict__ z, const float* __restrict__ msk,
                       const unsigned short* __restrict__ abuf, const unsigned short* __restrict__ bbuf,
                       const unsigned short* __restrict__ wpbf, const float* __restrict__ obias,
                       float* __restrict__ zout) {
  __shared__ unsigned short opmbf[16 * 1024];
  __shared__ float recip[16];
  int it = blockIdx.x / BN;      // 24 i-tiles
  int j  = blockIdx.x % BN;
  int i0 = it * 16;
  int t = threadIdx.x;           // 512 = 16 waves
  if (t < 16) {
    float num = 0.f;
    for (int s = 0; s < SS; s++) num += msk[s * BN + i0 + t] * msk[s * BN + j];
    recip[t] = __builtin_amdgcn_rcpf(fmaxf(num, 1.0f));
  }
  __syncthreads();
  int wave = t >> 5, lane = t & 31, mh = (lane >> 4) * 8;
  const unsigned short* A  = abuf + (long)i0 * COPM * SS;  // 512 rows (i_loc*32+c), stride SS
  const unsigned short* Bm = bbuf + (long)j * COPM * SS;   // 32 rows (d), stride SS
#pragma unroll
  for (int u = 0; u < 4; u++) {                            // 64 tiles over 16 waves
    int tile = wave * 4 + u;
    int mt = tile >> 1, nt = tile & 1;
    v8f acc = {};
    v16bf a = load_fragA(A, SS, mt * 16, 0, lane);
    v16bf b = load_fragB(Bm, SS, nt * 16, 0, lane);
    for (int kk = 32; kk < SS; kk += 32) {
      v16bf an = load_fragA(A, SS, mt * 16, kk, lane);
      v16bf bn = load_fragB(Bm, SS, nt * 16, kk, lane);
      acc = wmma_bf(a, b, acc);
      a = an; b = bn;
    }
    acc = wmma_bf(a, b, acc);
    int d = nt * 16 + (lane & 15);
#pragma unroll
    for (int r = 0; r < 8; r++) {
      int mrow = mt * 16 + mh + r;           // i_loc*32 + c
      int il = mrow >> 5, c = mrow & 31;
      opmbf[il * 1024 + c * 32 + d] = f2bfu(acc[r] * recip[il]);
    }
  }
  __syncthreads();
  if (wave < 8) {                             // projection: M=16, N=128, K=1024
    v8f acc = {};
    v16bf a = load_fragA(opmbf, 1024, 0, 0, lane);
    v16bf b = load_fragB(wpbf, 1024, wave * 16, 0, lane);
    for (int kk = 32; kk < 1024; kk += 32) {
      v16bf an = load_fragA(opmbf, 1024, 0, kk, lane);
      v16bf bn = load_fragB(wpbf, 1024, wave * 16, kk, lane);
      acc = wmma_bf(a, b, acc);
      a = an; b = bn;
    }
    acc = wmma_bf(a, b, acc);
    int zc = wave * 16 + (lane & 15);
#pragma unroll
    for (int r = 0; r < 8; r++) {
      int il = mh + r;
      long idx = ((long)(i0 + il) * BN + j) * ZD + zc;
      zout[idx] = z[idx] + acc[r] + obias[zc];
    }
  }
}

extern "C" void kernel_launch(void* const* d_in, const int* in_sizes, int n_in,
                              void* d_out, int out_size, void* d_ws, size_t ws_size,
                              hipStream_t stream) {
  const float* z     = (const float*)d_in[0];
  const float* m     = (const float*)d_in[1];
  const float* tmask = (const float*)d_in[2];
  const float* mmask = (const float*)d_in[3];
  const float* nzw = (const float*)d_in[4];
  const float* nzb = (const float*)d_in[5];
  const float* pzw = (const float*)d_in[6];
  const float* nmw = (const float*)d_in[7];
  const float* nmb = (const float*)d_in[8];
  const float* pmw = (const float*)d_in[9];
  const float* pgw = (const float*)d_in[10];
  const float* pow_ = (const float*)d_in[11];
  const float* tnw = (const float*)d_in[12];
  const float* tnb = (const float*)d_in[13];
  const float* f1  = (const float*)d_in[14];
  const float* f2  = (const float*)d_in[15];
  const float* f3  = (const float*)d_in[16];
  const float* onw = (const float*)d_in[17];
  const float* onb = (const float*)d_in[18];
  const float* paw = (const float*)d_in[19];
  const float* pbw = (const float*)d_in[20];
  const float* opw = (const float*)d_in[21];
  const float* opb = (const float*)d_in[22];

  unsigned char* ws = (unsigned char*)d_ws;
  size_t off = 0;
  float*          WBIAS = (float*)(ws + off);          off += (size_t)HH * BN * BN * 4;
  unsigned short* WBF   = (unsigned short*)(ws + off); off += (size_t)HH * BN * BN * 2;
  unsigned short* VBF   = (unsigned short*)(ws + off); off += (size_t)HH * SS * CH * BN * 2;
  unsigned short* GBF   = (unsigned short*)(ws + off); off += (size_t)SS * BN * HC * 2;
  unsigned short* GOBF  = (unsigned short*)(ws + off); off += (size_t)SS * BN * HC * 2;
  float*          M1    = (float*)(ws + off);          off += (size_t)SS * BN * MS * 4;
  unsigned short* ABUF  = (unsigned short*)(ws + off); off += (size_t)BN * COPM * SS * 2;
  unsigned short* BBUF  = (unsigned short*)(ws + off); off += (size_t)BN * COPM * SS * 2;
  unsigned short* WMBF  = (unsigned short*)(ws + off); off += 256 * 64 * 2;
  unsigned short* WGBF  = (unsigned short*)(ws + off); off += 256 * 64 * 2;
  unsigned short* WOBF  = (unsigned short*)(ws + off); off += 64 * 256 * 2;
  unsigned short* FC1B  = (unsigned short*)(ws + off); off += 256 * 64 * 2;
  unsigned short* FC2B  = (unsigned short*)(ws + off); off += 256 * 64 * 2;
  unsigned short* FC3B  = (unsigned short*)(ws + off); off += 64 * 256 * 2;
  unsigned short* WABF  = (unsigned short*)(ws + off); off += 32 * 64 * 2;
  unsigned short* WBBF  = (unsigned short*)(ws + off); off += 32 * 64 * 2;
  unsigned short* WPBF  = (unsigned short*)(ws + off); off += 128 * 1024 * 2;

  float* zout = (float*)d_out;
  float* mout = (float*)d_out + (size_t)BN * BN * ZD;

  auto cvt = [&](const float* s, unsigned short* d, int n) {
    k_cvt<<<(n + 255) / 256, 256, 0, stream>>>(s, d, n);
  };
  cvt(pmw, WMBF, 256 * 64);
  cvt(pgw, WGBF, 256 * 64);
  cvt(pow_, WOBF, 64 * 256);
  cvt(f1, FC1B, 256 * 64);
  cvt(f2, FC2B, 256 * 64);
  cvt(f3, FC3B, 64 * 256);
  cvt(paw, WABF, 32 * 64);
  cvt(pbw, WBBF, 32 * 64);
  cvt(opw, WPBF, 128 * 1024);

  k_bias<<<BN * BN, 128, 0, stream>>>(z, tmask, nzw, nzb, pzw, WBIAS);
  k_softmax<<<HH * BN, 128, 0, stream>>>(WBIAS, WBF);
  k2_mvg<<<SS * BN / 16, 128, 0, stream>>>(m, nmw, nmb, WMBF, WGBF, VBF, GBF);
  k3_pwa<<<(HH * 24 * 128) / 8, 256, 0, stream>>>(WBF, VBF, GBF, GOBF);
  k4_m1<<<SS * BN / 16, 128, 0, stream>>>(m, GOBF, WOBF, M1);
  k5_trans<<<SS * BN / 16, 128, 0, stream>>>(M1, tnw, tnb, FC1B, FC2B, FC3B, mout);
  k6_ab<<<SS * BN / 16, 64, 0, stream>>>(mout, mmask, onw, onb, WABF, WBBF, ABUF, BBUF);
  k7_opm<<<24 * BN, 512, 0, stream>>>(z, mmask, ABUF, BBUF, WPBF, opb, zout);

  (void)in_sizes; (void)n_in; (void)out_size; (void)ws_size;
}
